// ADT_Encoder_80865644249256
// MI455X (gfx1250) — matrix operand order repaired and checked by reading, not compiled, not run on hardware
//
#include <hip/hip_runtime.h>
#include <hip/hip_bf16.h>
#include <math.h>

typedef __attribute__((ext_vector_type(16))) __bf16 v16bf;
typedef __attribute__((ext_vector_type(8)))  float  v8f;

#define B_SZ        32
#define ADT_TOKENS  2048
#define ADT_COMP    2048
#define EMB         128
#define NLAYER      2
#define D_IN        256
#define N_ST        16
#define K_CONV      4
#define DT_R        8
#define REMAIN      1740           // int(2048 * 0.85)
#define SEQ         (REMAIN + 1)   // 1741 (cls + kept tokens)
#define MROWS       (B_SZ * SEQ)   // 55712, divisible by 16
#define XZ_LD       (2 * D_IN)     // 512
#define XDBC_LD     48             // 40 cols padded to 48
#define DTW_KPAD    32             // dt_w K padded 8 -> 32 with zeros

__device__ __forceinline__ void pack8(v16bf& v, int base, float4 p, float4 q)
{
    v[base + 0] = (__bf16)p.x; v[base + 1] = (__bf16)p.y;
    v[base + 2] = (__bf16)p.z; v[base + 3] = (__bf16)p.w;
    v[base + 4] = (__bf16)q.x; v[base + 5] = (__bf16)q.y;
    v[base + 6] = (__bf16)q.z; v[base + 7] = (__bf16)q.w;
}

// ---------------------------------------------------------------------------
// bf16 WMMA GEMM: C[m,n] = act( sum_k A[m,k]*B[n,k] + bias[n] )   (C = A @ B^T)
// One wave computes a 16 x (16*NSTRIP) strip: one A fragment is reused for
// NSTRIP v_wmma_f32_16x16x32_bf16 issues. K must be a multiple of 32.
// All loads are unconditional float4 (row pointers clamped; out-of-range
// results are simply not stored), so EXEC is all-ones throughout.
// act: 0 = none, 1 = softplus
// ---------------------------------------------------------------------------
template <int NSTRIP>
__global__ __launch_bounds__(256)
void gemm_bf16_wmma(const float* __restrict__ A, const float* __restrict__ Bm,
                    const float* __restrict__ bias, float* __restrict__ C,
                    int M, int N, int K, int lda, int ldb, int ldc, int act)
{
    const int nt    = (N + 15) >> 4;
    const int mt    = (M + 15) >> 4;
    const int njobs = (nt + NSTRIP - 1) / NSTRIP;
    const int job   = blockIdx.x * (blockDim.x >> 5) + (threadIdx.x >> 5);
    if (job >= mt * njobs) return;             // wave-uniform exit

    const int tm   = job / njobs;
    const int tn0  = (job - tm * njobs) * NSTRIP;
    const int lane = threadIdx.x & 31;
    const int hl   = lane >> 4;                // half-wave select
    const int l15  = lane & 15;

    int arow = tm * 16 + l15;                  // clamp -> safe address, result
    if (arow >= M) arow = M - 1;               // discarded by the store guard
    const float* Ap = A + (size_t)arow * lda;

    const float* Bp[NSTRIP];
#pragma unroll
    for (int t = 0; t < NSTRIP; ++t) {
        int brow = (tn0 + t) * 16 + l15;
        if (brow >= N) brow = N - 1;
        Bp[t] = Bm + (size_t)brow * ldb;
    }

    v8f acc[NSTRIP] = {};
    for (int k0 = 0; k0 < K; k0 += 32) {
        __builtin_prefetch(Ap + k0 + 128, 0, 3);     // global_prefetch_b8
        // A 16x32 bf16 layout: lanes 0-15 K=0..7 & 16..23, lanes 16-31 K=8..15 & 24..31
        const float4* a4lo = (const float4*)(Ap + k0 + hl * 8);
        const float4* a4hi = (const float4*)(Ap + k0 + 16 + hl * 8);
        const float4 a0 = a4lo[0], a1 = a4lo[1];
        const float4 a2 = a4hi[0], a3 = a4hi[1];
        v16bf af;
        pack8(af, 0, a0, a1);
        pack8(af, 8, a2, a3);
#pragma unroll
        for (int t = 0; t < NSTRIP; ++t) {
            // B 32x16 bf16 layout: lanes 0-15 K=0..7 & 16..23, lanes 16-31 K=8..15 & 24..31
            const float4* b4lo = (const float4*)(Bp[t] + k0 + hl * 8);
            const float4* b4hi = (const float4*)(Bp[t] + k0 + 16 + hl * 8);
            const float4 b0 = b4lo[0], b1 = b4lo[1], b2 = b4hi[0], b3 = b4hi[1];
            v16bf bfr;
            pack8(bfr, 0, b0, b1);
            pack8(bfr, 8, b2, b3);
            acc[t] = __builtin_amdgcn_wmma_f32_16x16x32_bf16(
                         false, af, false, bfr, (short)0, acc[t], false, false);
        }
    }

#pragma unroll
    for (int t = 0; t < NSTRIP; ++t) {
        const int ncol = (tn0 + t) * 16 + l15;
        if (ncol < N) {
            const float bv = bias ? bias[ncol] : 0.0f;
#pragma unroll
            for (int r = 0; r < 8; ++r) {      // C/D layout: VGPR r -> M = r + 8*half
                const int mrow = tm * 16 + r + hl * 8;
                if (mrow < M) {
                    float v = acc[t][r] + bv;
                    if (act == 1) v = (v > 20.f) ? v : log1pf(__expf(v));
                    C[(size_t)mrow * ldc + ncol] = v;
                }
            }
        }
    }
}

// zero-pad dt_w (256x8) -> (256x32) so the dt GEMM needs no K masking
__global__ void pad_dtw(const float* __restrict__ dw, float* __restrict__ out)
{
    const int gid = blockIdx.x * blockDim.x + threadIdx.x;
    if (gid >= D_IN * DTW_KPAD) return;
    const int k = gid & (DTW_KPAD - 1);
    const int n = gid >> 5;
    out[gid] = (k < DT_R) ? dw[n * DT_R + k] : 0.0f;
}

// ---------------------------------------------------------------------------
// Token embed + shuffle + cls prepend (rows m = b*SEQ + s, E=128 row-major)
// ---------------------------------------------------------------------------
__global__ void embed_gather(const float* __restrict__ p0,
                             const float* __restrict__ emb_w,
                             const float* __restrict__ emb_b,
                             const float* __restrict__ cls,
                             const float* __restrict__ pos,
                             const int*   __restrict__ fwd_idx,
                             float* __restrict__ x)
{
    const int gid = blockIdx.x * blockDim.x + threadIdx.x;
    if (gid >= MROWS * EMB) return;
    const int e = gid & (EMB - 1);
    const int m = gid >> 7;
    const int b = m / SEQ;
    const int s = m - b * SEQ;
    float v;
    if (s == 0) {
        v = cls[e];
    } else {
        const int t = fwd_idx[(s - 1) * B_SZ + b];
        v = p0[b * ADT_TOKENS + t] * emb_w[e] + emb_b[e] + pos[t * EMB + e];
    }
    x[gid] = v;
}

// backward_indexes = argsort(fwd_idx, axis=0) == inverse permutation scatter
__global__ void bwd_index(const int* __restrict__ fwd, int* __restrict__ out)
{
    const int gid = blockIdx.x * blockDim.x + threadIdx.x;
    if (gid >= ADT_COMP * B_SZ) return;
    const int s = gid >> 5;
    const int b = gid & 31;
    out[fwd[gid] * B_SZ + b] = s;
}

// depthwise causal conv (K=4, left pad 3) over xi = xz[:, 0:256], then SiLU
__global__ void conv_silu(const float* __restrict__ xz,
                          const float* __restrict__ cw,
                          const float* __restrict__ cb,
                          float* __restrict__ out)
{
    const int gid = blockIdx.x * blockDim.x + threadIdx.x;
    if (gid >= MROWS * D_IN) return;
    const int d = gid & (D_IN - 1);
    const int m = gid >> 8;
    const int b = m / SEQ;
    const int s = m - b * SEQ;
    float acc = cb[d];
#pragma unroll
    for (int k = 0; k < K_CONV; ++k) {
        const int sp = s - (K_CONV - 1) + k;
        if (sp >= 0)
            acc += xz[(size_t)(b * SEQ + sp) * XZ_LD + d] * cw[d * K_CONV + k];
    }
    out[gid] = acc / (1.0f + __expf(-acc));
}

// ---------------------------------------------------------------------------
// Selective scan: thread = (b, d, state-quad); 4 lanes share (b,d), 4 states
// each; y reduced via shfl_xor. y = (sum h*C + x*D) * silu(z) -> xz[:, 0:256]
// ---------------------------------------------------------------------------
__global__ __launch_bounds__(256)
void sel_scan(const float* __restrict__ dtb,
              const float* __restrict__ xic,
              float*       __restrict__ xz,
              const float* __restrict__ xdbc,
              const float* __restrict__ A_log,
              const float* __restrict__ D_res)
{
    const int gid = blockIdx.x * blockDim.x + threadIdx.x;
    const int ng  = gid & 3;
    const int d   = (gid >> 2) & (D_IN - 1);
    const int b   = gid >> 10;

    float Aneg[4], h[4] = {0.f, 0.f, 0.f, 0.f};
#pragma unroll
    for (int i = 0; i < 4; ++i) Aneg[i] = -__expf(A_log[d * N_ST + ng * 4 + i]);
    const float Dres = D_res[d];

    for (int s = 0; s < SEQ; ++s) {
        const size_t m  = (size_t)b * SEQ + s;
        const float dtv = dtb[m * D_IN + d];
        const float xv  = xic[m * D_IN + d];
        const float* bc = xdbc + m * XDBC_LD;
        const float dx  = dtv * xv;
        float y = 0.f;
#pragma unroll
        for (int i = 0; i < 4; ++i) {
            const int n = ng * 4 + i;
            const float dA = __expf(dtv * Aneg[i]);
            h[i] = dA * h[i] + dx * bc[DT_R + n];
            y += h[i] * bc[DT_R + N_ST + n];
        }
        y += __shfl_xor(y, 1, 32);
        y += __shfl_xor(y, 2, 32);
        if (ng == 0) {
            const float zv = xz[m * XZ_LD + D_IN + d];
            xz[m * XZ_LD + d] = (y + xv * Dres) * (zv / (1.0f + __expf(-zv)));
        }
    }
}

// LayerNorm over E=128, wave per row; output transposed to (T, B, E)
__global__ __launch_bounds__(256)
void layernorm_out(const float* __restrict__ x,
                   const float* __restrict__ lw,
                   const float* __restrict__ lb,
                   float* __restrict__ out)
{
    const int wid = blockIdx.x * 8 + (threadIdx.x >> 5);
    if (wid >= MROWS) return;
    const int lane = threadIdx.x & 31;
    const int b = wid / SEQ;
    const int s = wid - b * SEQ;
    const float* row = x + (size_t)wid * EMB;

    float v[4];
    float sum = 0.f;
#pragma unroll
    for (int i = 0; i < 4; ++i) { v[i] = row[lane + 32 * i]; sum += v[i]; }
#pragma unroll
    for (int o = 16; o >= 1; o >>= 1) sum += __shfl_xor(sum, o, 32);
    const float mu = sum * (1.0f / EMB);

    float var = 0.f;
#pragma unroll
    for (int i = 0; i < 4; ++i) { const float dv = v[i] - mu; var += dv * dv; }
#pragma unroll
    for (int o = 16; o >= 1; o >>= 1) var += __shfl_xor(var, o, 32);
    const float inv = rsqrtf(var * (1.0f / EMB) + 1e-5f);

    float* orow = out + ((size_t)s * B_SZ + b) * EMB;
#pragma unroll
    for (int i = 0; i < 4; ++i) {
        const int e = lane + 32 * i;
        orow[e] = (v[i] - mu) * inv * lw[e] + lb[e];
    }
}

// ---------------------------------------------------------------------------
static inline int gemm_blocks(int M, int N, int nstrip) {
    const int nt = (N + 15) >> 4;
    const int jobs = ((M + 15) >> 4) * ((nt + nstrip - 1) / nstrip);
    return (jobs + 7) / 8;                      // 8 waves per 256-thread block
}

extern "C" void kernel_launch(void* const* d_in, const int* in_sizes, int n_in,
                              void* d_out, int out_size, void* d_ws, size_t ws_size,
                              hipStream_t stream)
{
    const float* patches = (const float*)d_in[0];
    const float* tok_w   = (const float*)d_in[1];
    const float* tok_b   = (const float*)d_in[2];
    const float* emb_w   = (const float*)d_in[3];
    const float* emb_b   = (const float*)d_in[4];
    const float* cls     = (const float*)d_in[5];
    const float* pos     = (const float*)d_in[6];
    const int*   fwd     = (const int*)  d_in[7];
    const float* in_w    = (const float*)d_in[8];
    const float* conv_w  = (const float*)d_in[9];
    const float* conv_b  = (const float*)d_in[10];
    const float* xproj_w = (const float*)d_in[11];
    const float* dt_w    = (const float*)d_in[12];
    const float* dt_b    = (const float*)d_in[13];
    const float* A_log   = (const float*)d_in[14];
    const float* D_res   = (const float*)d_in[15];
    const float* out_w   = (const float*)d_in[16];
    const float* ln_w    = (const float*)d_in[17];
    const float* ln_b    = (const float*)d_in[18];

    char* ws = (char*)d_ws;
    size_t off = 0;
    auto walloc = [&](size_t bytes) -> void* {
        void* p = ws + off;
        off += (bytes + 255) & ~(size_t)255;
        return p;
    };
    float* p0    = (float*)walloc((size_t)B_SZ * ADT_TOKENS * 4);
    float* x     = (float*)walloc((size_t)MROWS * EMB      * 4);
    float* xz    = (float*)walloc((size_t)MROWS * XZ_LD    * 4);
    float* xic   = (float*)walloc((size_t)MROWS * D_IN     * 4);
    float* xdbc  = (float*)walloc((size_t)MROWS * XDBC_LD  * 4);
    float* dtb   = (float*)walloc((size_t)MROWS * D_IN     * 4);
    float* dtwp  = (float*)walloc((size_t)D_IN * DTW_KPAD  * 4);

    // 1) p0 = patches @ tok_w^T + tok_b      (32 x 2048 x 2048)
    gemm_bf16_wmma<4><<<gemm_blocks(B_SZ, ADT_TOKENS, 4), 256, 0, stream>>>(
        patches, tok_w, tok_b, p0,
        B_SZ, ADT_TOKENS, ADT_TOKENS, ADT_TOKENS, ADT_TOKENS, ADT_TOKENS, 0);

    // 2) embed + pos + shuffle + cls
    embed_gather<<<(MROWS * EMB + 255) / 256, 256, 0, stream>>>(
        p0, emb_w, emb_b, cls, pos, fwd, x);

    // 3) backward indexes into tail of d_out
    int* out_bwd = (int*)((float*)d_out + (size_t)MROWS * EMB);
    bwd_index<<<(ADT_COMP * B_SZ + 255) / 256, 256, 0, stream>>>(fwd, out_bwd);

    // 4) Mamba layers
    for (int l = 0; l < NLAYER; ++l) {
        const float* inw = in_w    + (size_t)l * 2 * D_IN * EMB;
        const float* cw  = conv_w  + (size_t)l * D_IN * K_CONV;
        const float* cb  = conv_b  + (size_t)l * D_IN;
        const float* xpw = xproj_w + (size_t)l * (DT_R + 2 * N_ST) * D_IN;
        const float* dw  = dt_w    + (size_t)l * D_IN * DT_R;
        const float* db  = dt_b    + (size_t)l * D_IN;
        const float* Al  = A_log   + (size_t)l * D_IN * N_ST;
        const float* Dl  = D_res   + (size_t)l * D_IN;
        const float* ow  = out_w   + (size_t)l * EMB * D_IN;

        // zero-padded dt_w for mask-free K=32 GEMM
        pad_dtw<<<(D_IN * DTW_KPAD + 255) / 256, 256, 0, stream>>>(dw, dtwp);

        // xz = x @ in_w^T            (55712 x 512 x 128)
        gemm_bf16_wmma<4><<<gemm_blocks(MROWS, XZ_LD, 4), 256, 0, stream>>>(
            x, inw, nullptr, xz, MROWS, XZ_LD, EMB, EMB, EMB, XZ_LD, 0);

        // xi = silu(causal depthwise conv(xz[:, :256]))
        conv_silu<<<(MROWS * D_IN + 255) / 256, 256, 0, stream>>>(xz, cw, cb, xic);

        // xdbc = xi @ xproj_w^T      (55712 x 40 x 256), 48-col padded, 3 N-tiles
        gemm_bf16_wmma<3><<<gemm_blocks(MROWS, DT_R + 2 * N_ST, 3), 256, 0, stream>>>(
            xic, xpw, nullptr, xdbc,
            MROWS, DT_R + 2 * N_ST, D_IN, D_IN, D_IN, XDBC_LD, 0);

        // dt = softplus(dt_r @ dtw_pad^T + dt_b)  (55712 x 256 x 32, zero-padded K)
        gemm_bf16_wmma<4><<<gemm_blocks(MROWS, D_IN, 4), 256, 0, stream>>>(
            xdbc, dtwp, db, dtb, MROWS, D_IN, DTW_KPAD, XDBC_LD, DTW_KPAD, D_IN, 1);

        // selective scan + D residual + silu(z) gate -> y in xz[:, :256]
        sel_scan<<<(B_SZ * D_IN * 4) / 256, 256, 0, stream>>>(
            dtb, xic, xz, xdbc, Al, Dl);

        // x = y @ out_w^T            (55712 x 128 x 256), A stride 512
        gemm_bf16_wmma<4><<<gemm_blocks(MROWS, EMB, 4), 256, 0, stream>>>(
            xz, ow, nullptr, x, MROWS, EMB, D_IN, XZ_LD, D_IN, EMB, 0);
    }

    // 5) LayerNorm -> features (T, B, E) at head of d_out
    layernorm_out<<<(MROWS + 7) / 8, 256, 0, stream>>>(x, ln_w, ln_b, (float*)d_out);
}
